// ViewEncoderLayer_10058813407316
// MI455X (gfx1250) — matrix operand order repaired and checked by reading, not compiled, not run on hardware
//
#include <hip/hip_runtime.h>
#include <hip/hip_bf16.h>

typedef unsigned short u16;
typedef unsigned int u32;
typedef unsigned long long u64;
typedef __attribute__((ext_vector_type(16))) __bf16 v16bf;
typedef __attribute__((ext_vector_type(8))) float v8f;
typedef __attribute__((ext_vector_type(4))) u32 u32x4;
typedef __attribute__((ext_vector_type(8))) int i32x8;
typedef __attribute__((ext_vector_type(4))) int i32x4;

#ifndef __has_builtin
#define __has_builtin(x) 0
#endif
#if __has_builtin(__builtin_amdgcn_tensor_load_to_lds) && \
    __has_builtin(__builtin_amdgcn_s_wait_tensorcnt)
#define USE_TDM 1
#else
#define USE_TDM 0
#endif

namespace {

constexpr int kN = 4, kL = 8192, kS = 8192, kD = 256, kH = 8, kDH = 32;
constexpr int kNL = kN * kL;
constexpr size_t MB16 = 16u * 1024u * 1024u;

__device__ __forceinline__ u16 f2bf(float f) {
  u32 u = __builtin_bit_cast(u32, f);
  return (u16)((u + 0x7FFFu + ((u >> 16) & 1u)) >> 16);
}
__device__ __forceinline__ float bf2f(u16 h) {
  u32 u = (u32)h << 16;
  return __builtin_bit_cast(float, u);
}

union Frag {
  u32 u[8];
  v16bf v;
};

#if USE_TDM
// ---------------------------------------------------------------------------
// Issue one TDM 2D tile load (bf16 elements) global -> LDS.
// D# packed per CDNA5 ISA 8.3-8.5: group0 = {count/flags, lds_addr,
// global_addr, type=2}; group1 = dims/strides (fields are not dword aligned).
// pad=true inserts 1 DWORD of LDS padding every 16 DWORDs (64B rows -> 68B
// stride, bank-conflict-free A-tile rows).
// 6-arg toolchain variant: (g0, g1, g2, g3, g4, cpol).
// ---------------------------------------------------------------------------
__device__ __forceinline__ void tdm_load_2d(u32 lds_addr, const void* gptr,
                                            u32 tile_d0, u32 tile_d1,
                                            u64 stride0_elems, bool pad) {
  const u64 ga = (u64)gptr;
  u32x4 g0;
  g0[0] = 1u;                                            // count=1, user mode
  g0[1] = lds_addr;                                      // LDS byte address
  g0[2] = (u32)ga;                                       // global addr [31:0]
  g0[3] = (u32)((ga >> 32) & 0x1FFFFFFu) | (2u << 30);   // addr[56:32] | type=2
  u32 w0 = (1u << 16);                                   // data_size=1 -> 2B
  if (pad) w0 |= (1u << 20) | (3u << 22);                // pad_en, interval=16DW, amount=1DW
  i32x8 g1;
  g1[0] = (int)w0;
  g1[1] = (int)((tile_d0 & 0xFFFFu) << 16);                         // tensor_dim0 = tile_d0
  g1[2] = (int)((tile_d0 >> 16) | ((tile_d1 & 0xFFFFu) << 16));     // tensor_dim1 = tile_d1
  g1[3] = (int)((tile_d1 >> 16) | (tile_d0 << 16));                 // tile_dim0
  g1[4] = (int)tile_d1;                                             // tile_dim1 (tile_dim2=0)
  g1[5] = (int)(u32)(stride0_elems & 0xFFFFFFFFu);                  // dim0 stride lo
  g1[6] = (int)(u32)((stride0_elems >> 32) & 0xFFFFu);              // stride hi (stride1=0)
  g1[7] = 0;
  const i32x4 z4 = {0, 0, 0, 0};
  const i32x8 z8 = {0, 0, 0, 0, 0, 0, 0, 0};
  __builtin_amdgcn_tensor_load_to_lds(g0, g1, z4, z4, z8, 0);
}
#endif

// ---------------------------------------------------------------------------
// f32 -> bf16 conversion (grid-stride)
// ---------------------------------------------------------------------------
__global__ void cvt_kernel(const float* __restrict__ s, u16* __restrict__ d, int n) {
  for (int i = blockIdx.x * blockDim.x + threadIdx.x; i < n; i += gridDim.x * blockDim.x)
    d[i] = f2bf(s[i]);
}

// x (f32) -> left half of concat buffer (bf16, ldc=512)
__global__ void cat_x_kernel(const float* __restrict__ x, u16* __restrict__ catb) {
  size_t i = (size_t)blockIdx.x * 256 + threadIdx.x;
  size_t row = i >> 8;
  int col = (int)(i & 255);
  catb[row * 512 + col] = f2bf(x[i]);
}

// ---------------------------------------------------------------------------
// Tiled bf16 GEMM: C[M x Ncols] = act(A[M x K] @ B[K x Ncols])
// 128x128 block tile, 8 waves, each wave 32x64 (2x4 WMMA 16x16x32 bf16).
// Tiles are DMA'd into LDS by the Tensor Data Mover (wave 0 issues, TENSORcnt
// ordered, double buffered: tile i+1 in flight while tile i computes).
// ACT: 0=none, 1=elu+1, 2=LeakyReLU(0.1)
// ---------------------------------------------------------------------------
template <int ACT, bool OUT_BF16>
__global__ __launch_bounds__(256) void gemm_bf16_kernel(
    const u16* __restrict__ A, const u16* __restrict__ B, void* __restrict__ Cv,
    int Ncols, int K, int ldc, int coff) {
  __shared__ u16 As[2][128][34];   // [row][k]; TDM pad gives the 34-stride
  __shared__ u16 Bs[2][32][128];   // [k][n] row-major
  const int t = threadIdx.x;
  const int lane = t & 31, wave = t >> 5;
  const int wm = wave & 3, wn = wave >> 2;
  const int lm = lane & 15, lg = lane >> 4;
  const size_t row0 = (size_t)blockIdx.y * 128;
  const int col0 = blockIdx.x * 128;
  const int nk = K / 32;

  const v8f vzero = {0.f, 0.f, 0.f, 0.f, 0.f, 0.f, 0.f, 0.f};
  v8f acc[2][4];
#pragma unroll
  for (int i = 0; i < 2; ++i)
#pragma unroll
    for (int j = 0; j < 4; ++j) acc[i][j] = vzero;

#if USE_TDM
  if (wave == 0) {
    tdm_load_2d((u32)(uintptr_t)&As[0][0][0], A + row0 * (size_t)K, 32, 128,
                (u64)K, true);
    tdm_load_2d((u32)(uintptr_t)&Bs[0][0][0], B + col0, 128, 32, (u64)Ncols,
                false);
  }
#endif

  for (int ki = 0; ki < nk; ++ki) {
    const int cur = ki & 1;
#if USE_TDM
    if (wave == 0) {
      if (ki + 1 < nk) {
        tdm_load_2d((u32)(uintptr_t)&As[cur ^ 1][0][0],
                    A + row0 * (size_t)K + (ki + 1) * 32, 32, 128, (u64)K, true);
        tdm_load_2d((u32)(uintptr_t)&Bs[cur ^ 1][0][0],
                    B + (size_t)((ki + 1) * 32) * Ncols + col0, 128, 32,
                    (u64)Ncols, false);
        __builtin_amdgcn_s_wait_tensorcnt((short)2);  // tile ki landed
      } else {
        __builtin_amdgcn_s_wait_tensorcnt((short)0);
      }
    }
#else
    // Fallback: manual staging (all threads)
#pragma unroll
    for (int i = 0; i < 2; ++i) {
      int chunk = t * 2 + i;  // 0..511
      int r = chunk >> 2, c = (chunk & 3) * 8;
      const uint4 d = *(const uint4*)(A + (row0 + r) * (size_t)K + ki * 32 + c);
      u32* dp = (u32*)&As[cur][r][c];
      dp[0] = d.x; dp[1] = d.y; dp[2] = d.z; dp[3] = d.w;
    }
#pragma unroll
    for (int i = 0; i < 2; ++i) {
      int chunk = t * 2 + i;
      int kk = chunk >> 4, nn = (chunk & 15) * 8;
      const uint4 d = *(const uint4*)(B + (size_t)(ki * 32 + kk) * Ncols + col0 + nn);
      *(uint4*)&Bs[cur][kk][nn] = d;
    }
#endif
    __syncthreads();

    // A fragments: lane (m=lm, K half lg): VGPR j -> K=16*lg+2j
    Frag af[2];
#pragma unroll
    for (int mt = 0; mt < 2; ++mt) {
      const u32* rp = (const u32*)&As[cur][wm * 32 + mt * 16 + lm][0];
#pragma unroll
      for (int j = 0; j < 4; ++j) {
        af[mt].u[j] = rp[8 * lg + j];
        af[mt].u[4 + j] = rp[8 * lg + 4 + j];
      }
    }
    // B fragments: lane (n=lm, K half lg): VGPR j -> K=16*lg+2j
    Frag bq[4];
#pragma unroll
    for (int nt = 0; nt < 4; ++nt) {
      const u16* bp = &Bs[cur][lg * 16][wn * 64 + nt * 16 + lm];
#pragma unroll
      for (int j = 0; j < 8; ++j) {
        u32 lo = bp[(2 * j) * 128];
        u32 hi = bp[(2 * j + 1) * 128];
        bq[nt].u[j] = lo | (hi << 16);
      }
    }
#pragma unroll
    for (int mt = 0; mt < 2; ++mt)
#pragma unroll
      for (int nt = 0; nt < 4; ++nt)
        acc[mt][nt] = __builtin_amdgcn_wmma_f32_16x16x32_bf16(
            false, af[mt].v, false, bq[nt].v, (short)0, acc[mt][nt], false, false);
    __syncthreads();  // safe to overwrite this buffer next round
  }

  // epilogue: D layout f32 16x16: m = r + 8*lg, n = lm
#pragma unroll
  for (int mt = 0; mt < 2; ++mt)
#pragma unroll
    for (int nt = 0; nt < 4; ++nt)
#pragma unroll
      for (int r = 0; r < 8; ++r) {
        float v = acc[mt][nt][r];
        if (ACT == 1) v = v > 0.f ? v + 1.f : __expf(v);  // elu(x)+1
        if (ACT == 2) v = v >= 0.f ? v : 0.1f * v;        // LeakyReLU(0.1)
        size_t row = row0 + wm * 32 + mt * 16 + r + 8 * lg;
        int col = col0 + wn * 64 + nt * 16 + lm;
        if (OUT_BF16)
          ((u16*)Cv)[row * (size_t)ldc + coff + col] = f2bf(v);
        else
          ((float*)Cv)[row * (size_t)ldc + coff + col] = v;
      }
}

// ---------------------------------------------------------------------------
// KV[n,h,d,v] = sum_s K[n,s,h,d] * V[n,s,h,v]  (32x32, K=8192 contraction)
// + Ksum[n,h,d] = sum_s K[n,s,h,d]
// One block per (n,h); 8 waves own s-slices; 2x2 WMMA fragments; cross-wave
// reduce via LDS ds_add_f32 atomics. KV written transposed (KVt[v][d], bf16).
// The /v_len and *v_len scalings cancel -> dropped.
// ---------------------------------------------------------------------------
__global__ __launch_bounds__(256) void kv_kernel(
    const u16* __restrict__ Kb, const u16* __restrict__ Vb,
    u16* __restrict__ KVt, float* __restrict__ Ksum) {
  __shared__ float kv[32][32];
  __shared__ float ks[32];
  const int t = threadIdx.x;
  const int lane = t & 31, wave = t >> 5;
  const int lm = lane & 15, lg = lane >> 4;
  const int nh = blockIdx.x;
  const int n = nh >> 3, h = nh & 7;

  for (int i = t; i < 1024; i += 256) kv[i >> 5][i & 31] = 0.f;
  if (t < 32) ks[t] = 0.f;
  __syncthreads();

  const size_t sstr = (size_t)kH * kDH;  // 256 elements between s steps
  const u16* Kbase = Kb + ((size_t)n * kS * kH + h) * kDH;
  const u16* Vbase = Vb + ((size_t)n * kS * kH + h) * kDH;

  const v8f vzero = {0.f, 0.f, 0.f, 0.f, 0.f, 0.f, 0.f, 0.f};
  v8f acc[2][2];
#pragma unroll
  for (int a = 0; a < 2; ++a)
#pragma unroll
    for (int b = 0; b < 2; ++b) acc[a][b] = vzero;

  for (int it = wave; it < kS / 32; it += 8) {
    const int s0 = it * 32;
    Frag af[2];  // A = K^T : A[m=d][k=s']
#pragma unroll
    for (int dt = 0; dt < 2; ++dt) {
      const int d = dt * 16 + lm;
      const int kb = lg * 16;
#pragma unroll
      for (int j = 0; j < 4; ++j) {
        int sa = s0 + kb + 2 * j;
        u32 lo = Kbase[(size_t)sa * sstr + d];
        u32 hi = Kbase[(size_t)(sa + 1) * sstr + d];
        af[dt].u[j] = lo | (hi << 16);
        int sb2 = s0 + 8 + kb + 2 * j;
        lo = Kbase[(size_t)sb2 * sstr + d];
        hi = Kbase[(size_t)(sb2 + 1) * sstr + d];
        af[dt].u[4 + j] = lo | (hi << 16);
      }
    }
    Frag bq[2];  // B = V : B[k=s'][n=v]
#pragma unroll
    for (int vt = 0; vt < 2; ++vt) {
      const int vcol = vt * 16 + lm;
      const int kb2 = lg * 16;
#pragma unroll
      for (int j = 0; j < 8; ++j) {
        int sv = s0 + kb2 + 2 * j;
        u32 lo = Vbase[(size_t)sv * sstr + vcol];
        u32 hi = Vbase[(size_t)(sv + 1) * sstr + vcol];
        bq[vt].u[j] = lo | (hi << 16);
      }
    }
#pragma unroll
    for (int dt = 0; dt < 2; ++dt)
#pragma unroll
      for (int vt = 0; vt < 2; ++vt)
        acc[dt][vt] = __builtin_amdgcn_wmma_f32_16x16x32_bf16(
            false, af[dt].v, false, bq[vt].v, (short)0, acc[dt][vt], false, false);
  }

  {  // Ksum partial sums (8 s-lanes per d)
    float s = 0.f;
    const int d = t & 31, sb0 = t >> 5;
    for (int si = sb0; si < kS; si += 8) s += bf2f(Kbase[(size_t)si * sstr + d]);
    atomicAdd(&ks[d], s);
  }
#pragma unroll
  for (int dt = 0; dt < 2; ++dt)
#pragma unroll
    for (int vt = 0; vt < 2; ++vt)
#pragma unroll
      for (int r = 0; r < 8; ++r)
        atomicAdd(&kv[dt * 16 + r + 8 * lg][vt * 16 + lm], acc[dt][vt][r]);
  __syncthreads();

  if (t < 32) Ksum[nh * 32 + t] = ks[t];
  for (int i = t; i < 1024; i += 256) {
    int vv = i >> 5, dd = i & 31;
    KVt[(size_t)nh * 1024 + i] = f2bf(kv[dd][vv]);  // KVt[v][d]
  }
}

// ---------------------------------------------------------------------------
// msg[n,l,h,v] = (Q[n,l,h,:] @ KV[n,h,:,v]) * Z[n,l,h],  Z = 1/(Q.Ksum + eps)
// ---------------------------------------------------------------------------
__global__ __launch_bounds__(256) void attn_kernel(
    const u16* __restrict__ Qb, const u16* __restrict__ KVt,
    const float* __restrict__ Ksum, u16* __restrict__ attnb) {
  __shared__ float zs[128];
  __shared__ float ks[32];
  const int t = threadIdx.x;
  const int lane = t & 31, wave = t >> 5;
  const int lm = lane & 15, lg = lane >> 4;
  const int l0 = blockIdx.x * 128;
  const int nh = blockIdx.y;
  const int n = nh >> 3, h = nh & 7;
  const u16* Qbase = Qb + ((size_t)n * kL * kH + h) * kDH;

  if (t < 32) ks[t] = Ksum[nh * 32 + t];
  __syncthreads();
  if (t < 128) {
    const u16* qp = Qbase + (size_t)(l0 + t) * (kH * kDH);
    float dot = 0.f;
#pragma unroll
    for (int d = 0; d < 32; ++d) dot += bf2f(qp[d]) * ks[d];
    zs[t] = 1.0f / (dot + 1e-6f);
  }
  __syncthreads();

  const int rowb = wave * 16;
  Frag af;
  {
    const u32* qp = (const u32*)(Qbase + (size_t)(l0 + rowb + lm) * (kH * kDH));
#pragma unroll
    for (int j = 0; j < 4; ++j) {
      af.u[j] = qp[8 * lg + j];
      af.u[4 + j] = qp[8 * lg + 4 + j];
    }
  }
  const v8f vzero = {0.f, 0.f, 0.f, 0.f, 0.f, 0.f, 0.f, 0.f};
#pragma unroll
  for (int nt = 0; nt < 2; ++nt) {
    Frag bq;
    const u32* rp = (const u32*)(KVt + (size_t)nh * 1024 + (size_t)(nt * 16 + lm) * 32);
#pragma unroll
    for (int j = 0; j < 8; ++j) bq.u[j] = rp[8 * lg + j];
    v8f c = __builtin_amdgcn_wmma_f32_16x16x32_bf16(
        false, af.v, false, bq.v, (short)0, vzero, false, false);
#pragma unroll
    for (int r = 0; r < 8; ++r) {
      int m = rowb + r + 8 * lg;
      float v = c[r] * zs[m];
      attnb[((size_t)n * kL + l0 + m) * kD + h * kDH + nt * 16 + lm] = f2bf(v);
    }
  }
}

// ---------------------------------------------------------------------------
// LayerNorm over D=256. RESID=false: bf16 out into cat (ldo=512, coff=256).
// RESID=true: f32 out = x + LN(h2).
// ---------------------------------------------------------------------------
template <bool RESID>
__global__ __launch_bounds__(256) void ln_kernel(
    const float* __restrict__ X, const float* __restrict__ gm,
    const float* __restrict__ bt, const float* __restrict__ resid,
    void* __restrict__ out, int ldo, int coff) {
  __shared__ float red[256];
  __shared__ float sm, sv;
  const int t = threadIdx.x;
  const size_t row = blockIdx.x;
  const float xv = X[row * kD + t];
  red[t] = xv;
  __syncthreads();
  for (int s = 128; s > 0; s >>= 1) {
    if (t < s) red[t] += red[t + s];
    __syncthreads();
  }
  if (t == 0) sm = red[0] * (1.0f / kD);
  __syncthreads();
  const float xc = xv - sm;
  red[t] = xc * xc;
  __syncthreads();
  for (int s = 128; s > 0; s >>= 1) {
    if (t < s) red[t] += red[t + s];
    __syncthreads();
  }
  if (t == 0) sv = red[0] * (1.0f / kD);
  __syncthreads();
  float y = xc * rsqrtf(sv + 1e-5f) * gm[t] + bt[t];
  if (RESID)
    ((float*)out)[row * (size_t)ldo + coff + t] = resid[row * kD + t] + y;
  else
    ((u16*)out)[row * (size_t)ldo + coff + t] = f2bf(y);
}

}  // namespace

extern "C" void kernel_launch(void* const* d_in, const int* in_sizes, int n_in,
                              void* d_out, int out_size, void* d_ws, size_t ws_size,
                              hipStream_t stream) {
  (void)in_sizes; (void)n_in; (void)out_size; (void)ws_size;
  const float* x   = (const float*)d_in[0];
  const float* src = (const float*)d_in[1];
  const float* Wq  = (const float*)d_in[2];
  const float* Wk  = (const float*)d_in[3];
  const float* Wv  = (const float*)d_in[4];
  const float* Wm  = (const float*)d_in[5];
  const float* W1  = (const float*)d_in[6];
  const float* W2  = (const float*)d_in[7];
  const float* g1  = (const float*)d_in[8];
  const float* b1  = (const float*)d_in[9];
  const float* g2  = (const float*)d_in[10];
  const float* b2  = (const float*)d_in[11];

  char* ws = (char*)d_ws;
  // Overlaid workspace (buffers reused once dead); peak ~86 MB.
  u16* Qb    = (u16*)(ws);
  u16* Kb    = (u16*)(ws + MB16);
  u16* Vb    = (u16*)(ws + 2 * MB16);
  u16* xb    = (u16*)(ws + 3 * MB16);
  u16* sb    = (u16*)(ws + 4 * MB16);
  u16* attnb = sb;                          // sb dead after V GEMM
  float* msgW = (float*)(ws + MB16);        // overlays Kb+Vb (dead after kv)
  u16* catb  = (u16*)(ws + 3 * MB16);       // overlays xb+sb (dead)
  u16* hb    = (u16*)(ws + MB16);           // overlays msgW (dead after LN1)
  float* h2  = (float*)(ws + 3 * MB16);     // overlays catb (dead after W1/W2)
  char* wp = ws + 5 * MB16;
  u16* Wqb = (u16*)(wp);
  u16* Wkb = (u16*)(wp + 131072);
  u16* Wvb = (u16*)(wp + 262144);
  u16* Wmb = (u16*)(wp + 393216);
  u16* W1b = (u16*)(wp + 524288);
  u16* W2b = (u16*)(wp + 1048576);
  u16* KVt = (u16*)(wp + 1310720);
  float* Ksum = (float*)(wp + 1376256);

  const dim3 thr(256);
  auto cvt = [&](const float* s, u16* d, int n) {
    int blocks = (n + 255) / 256;
    if (blocks > 8192) blocks = 8192;
    cvt_kernel<<<dim3(blocks), thr, 0, stream>>>(s, d, n);
  };
  cvt(x, xb, kNL * kD);
  cvt(src, sb, kNL * kD);
  cvt(Wq, Wqb, kD * kD);
  cvt(Wk, Wkb, kD * kD);
  cvt(Wv, Wvb, kD * kD);
  cvt(Wm, Wmb, kD * kD);
  cvt(W1, W1b, 4 * kD * kD);
  cvt(W2, W2b, 2 * kD * kD);

  const dim3 g256(kD / 128, kNL / 128);
  const dim3 g512(2 * kD / 128, kNL / 128);
  // Q = elu(x@Wq)+1 ; K = elu(src@Wk)+1 ; V = src@Wv
  gemm_bf16_kernel<1, true><<<g256, thr, 0, stream>>>(xb, Wqb, Qb, kD, kD, kD, 0);
  gemm_bf16_kernel<1, true><<<g256, thr, 0, stream>>>(sb, Wkb, Kb, kD, kD, kD, 0);
  gemm_bf16_kernel<0, true><<<g256, thr, 0, stream>>>(sb, Wvb, Vb, kD, kD, kD, 0);
  // KV + Ksum per (n,h)
  kv_kernel<<<dim3(kN * kH), thr, 0, stream>>>(Kb, Vb, KVt, Ksum);
  // msg = (Q@KV) * Z
  attn_kernel<<<dim3(kL / 128, kN * kH), thr, 0, stream>>>(Qb, KVt, Ksum, attnb);
  // msgW = msg @ Wm (f32)
  gemm_bf16_kernel<0, false><<<g256, thr, 0, stream>>>(attnb, Wmb, msgW, kD, kD, kD, 0);
  // cat = [bf16(x) | LN1(msgW)]
  cat_x_kernel<<<dim3(kNL * kD / 256), thr, 0, stream>>>(x, catb);
  ln_kernel<false><<<dim3(kNL), thr, 0, stream>>>(msgW, g1, b1, nullptr, catb, 2 * kD, kD);
  // h = LeakyReLU(cat @ W1) ; h2 = h @ W2 (f32)
  gemm_bf16_kernel<2, true><<<g512, thr, 0, stream>>>(catb, W1b, hb, 2 * kD, 2 * kD, 2 * kD, 0);
  gemm_bf16_kernel<0, false><<<g256, thr, 0, stream>>>(hb, W2b, h2, kD, 2 * kD, kD, 0);
  // out = x + LN2(h2)
  ln_kernel<true><<<dim3(kNL), thr, 0, stream>>>(h2, g2, b2, x, (float*)d_out, kD, 0);
}
